// comb_encoder_88691074662693
// MI455X (gfx1250) — compile-verified, matrix operand order checked by reading, not run on hardware
//
#include <hip/hip_runtime.h>
#include <hip/hip_bf16.h>
#include <stdint.h>

// Problem constants (fixed by the harness / reference).
#define T_LEN   8192
#define D_IN    512
#define H_DIM   768
#define G4      3072      // 4*H
#define X1_DIM  1536      // 2*H
#define E_CNT   16384
#define G_CNT   2048
#define S_CNT   32

// Recurrence partitioning: 16 cooperative workgroups, each owns H/16 = 48 h-lanes
// -> 192 gate rows; weights resident in LDS with padded row stride.
#define NWG_REC 16
#define HSLICE  48
#define GSLICE  192
#define KW32    384       // 768 bf16 = 384 dwords per gate row (source)
#define KWPAD   386       // padded LDS row stride (386%64==2 -> conflict-free wave32)

typedef __bf16 bf16_t;
typedef __attribute__((ext_vector_type(16))) __bf16 v16bf;
typedef __attribute__((ext_vector_type(8)))  float  v8f;
typedef __attribute__((ext_vector_type(4)))  unsigned int v4u;
typedef __attribute__((ext_vector_type(4)))  int v4i;

union V16U { v16bf bf; v4u u2[2]; };

// CDNA5 async global->LDS path (guarded: falls back to synchronous copies).
#if defined(__HIP_DEVICE_COMPILE__) && __has_builtin(__builtin_amdgcn_global_load_async_to_lds_b128)
#define HAVE_ASYNC_LDS 1
#else
#define HAVE_ASYNC_LDS 0
#endif

#if HAVE_ASYNC_LDS
__device__ __forceinline__ void wait_async_1() {
#if __has_builtin(__builtin_amdgcn_s_wait_asynccnt)
  __builtin_amdgcn_s_wait_asynccnt(1);
#else
  asm volatile("s_wait_asynccnt 0x1" ::: "memory");
#endif
}
__device__ __forceinline__ void wait_async_0() {
#if __has_builtin(__builtin_amdgcn_s_wait_asynccnt)
  __builtin_amdgcn_s_wait_asynccnt(0);
#else
  asm volatile("s_wait_asynccnt 0x0" ::: "memory");
#endif
}
#endif

__device__ __forceinline__ unsigned short f2bf(float f) {
  unsigned u = __float_as_uint(f);
  u += 0x7fffu + ((u >> 16) & 1u);       // round-to-nearest-even
  return (unsigned short)(u >> 16);
}
__device__ __forceinline__ float bf_lo(unsigned int u) { return __uint_as_float(u << 16); }
__device__ __forceinline__ float bf_hi(unsigned int u) { return __uint_as_float(u & 0xffff0000u); }

// ---------------------------------------------------------------- conversions
__global__ void cvt_f32_bf16(const float* __restrict__ src,
                             unsigned short* __restrict__ dst, int n) {
  int i = blockIdx.x * blockDim.x + threadIdx.x;
  if (i < n) dst[i] = f2bf(src[i]);
}

__global__ void zero_u32(unsigned int* p, int n) {
  int i = blockIdx.x * blockDim.x + threadIdx.x;
  if (i < n) p[i] = 0u;
}

// ------------------------------------------------------------------- WMMA GEMM
// C[M,N] = A[M,K](bf16 rm) * B[N,K]^T(bf16 rm-over-K) + bias[N].
// Block = 8 waves = 128x64 tile. B panel (64x32 per K-step) is staged in LDS
// (shared by all 8 waves, double buffered, async-to-LDS when available);
// A fragments stream straight from global per wave (pipelines against WMMA).
// LDS B row stride = 80B: bank stride 20 -> 16 lanes hit 16 distinct banks.
#define BROW 80
__global__ __launch_bounds__(256) void gemm_bf16_wmma(
    const unsigned short* __restrict__ A,
    const unsigned short* __restrict__ B,
    const float* __restrict__ bias,
    float* __restrict__ C,
    int M, int N, int K) {
  __shared__ __attribute__((aligned(16))) unsigned char ldsB[2][64 * BROW];

  const int tid = threadIdx.x;
  const int lane = tid & 31;
  const int wave = tid >> 5;
  const int tilesN = N >> 6;                   // N/64
  const int bm = blockIdx.x / tilesN;          // 128-row block
  const int tn = blockIdx.x - bm * tilesN;

  const int mrow = bm * 128 + wave * 16 + (lane & 15);
  const int koff = (lane >> 4) * 8;
  const int nloc = lane & 15;
  const int n0 = tn * 64;

  // Stage helper mapping: 256 threads <-> 256 16B chunks (64 rows x 4 chunks).
  const int srow = tid >> 2;
  const int spart = tid & 3;
  const unsigned short* __restrict__ sbase = B + (size_t)(n0 + srow) * K + spart * 8;

  auto stage = [&](int k0, int buf) {
    const unsigned short* src = sbase + k0;
    unsigned char* dst = &ldsB[buf][srow * BROW + spart * 16];
#if HAVE_ASYNC_LDS
    __builtin_amdgcn_global_load_async_to_lds_b128(
        (__attribute__((address_space(1))) v4i*)src,
        (__attribute__((address_space(3))) v4i*)dst, 0, 0);
#else
    *(v4u*)dst = *(const v4u*)src;
#endif
  };

  v8f acc[4] = {};
  const unsigned short* __restrict__ arow = A + (size_t)mrow * K;

  stage(0, 0);
  int p = 0;
  for (int k0 = 0; k0 < K; k0 += 32) {
    const bool more = (k0 + 32) < K;
    if (more) stage(k0 + 32, p ^ 1);
#if HAVE_ASYNC_LDS
    if (more) wait_async_1(); else wait_async_0();
#endif
    __syncthreads();

    V16U a;
    a.u2[0] = *(const v4u*)(arow + k0 + koff);
    a.u2[1] = *(const v4u*)(arow + k0 + koff + 16);
    __builtin_prefetch(arow + k0 + 64, 0, 1);      // global_prefetch_b8
#pragma unroll
    for (int jj = 0; jj < 4; ++jj) {
      const unsigned char* bp = &ldsB[p][(jj * 16 + nloc) * BROW + koff * 2];
      V16U b;
      b.u2[0] = *(const v4u*)bp;
      b.u2[1] = *(const v4u*)(bp + 32);
      acc[jj] = __builtin_amdgcn_wmma_f32_16x16x32_bf16(
          false, a.bf, false, b.bf, (short)0, acc[jj], false, false);
    }
    __syncthreads();     // all waves done reading buffer p before it is restaged
    p ^= 1;
  }

  // C/D layout: VGPR v -> M = v + 8*(lane>>4); N = lane&15.
  const int mbase = bm * 128 + wave * 16 + 8 * (lane >> 4);
#pragma unroll
  for (int jj = 0; jj < 4; ++jj) {
    const int col = n0 + jj * 16 + nloc;
    const float bv = bias[col];
#pragma unroll
    for (int v = 0; v < 8; ++v) {
      C[(size_t)(mbase + v) * N + col] = acc[jj][v] + bv;
    }
  }
}

// --------------------------------------------------------------- grid-wide sync
__device__ __forceinline__ void grid_sync(unsigned int* ctr, unsigned int* phase,
                                          int tid) {
  __threadfence();             // make this WG's h stores device-visible
  __syncthreads();
  unsigned int p = ++(*phase);
  if (tid == 0) {
    __hip_atomic_fetch_add(ctr, 1u, __ATOMIC_ACQ_REL, __HIP_MEMORY_SCOPE_AGENT);
    const unsigned int target = (unsigned int)NWG_REC * p;
    while (__hip_atomic_load(ctr, __ATOMIC_ACQUIRE, __HIP_MEMORY_SCOPE_AGENT) < target)
      __builtin_amdgcn_s_sleep(2);
  }
  __builtin_amdgcn_s_cluster_barrier();  // NOP unless dispatched as a cluster
  __syncthreads();
}

// ------------------------------------------------------------------ recurrence
// 16 persistent WGs x 256 threads. WG w owns h rows [w*48, w*48+48) and the 192
// corresponding gate rows; their weights stay resident in LDS (padded stride).
// h broadcast via global double buffer hg[2][768]; one grid sync per timestep.
__global__ __launch_bounds__(256) void lstm_recur(
    const float* __restrict__ pre,          // T x 3072 (x@w_ih.T + b)
    const unsigned int* __restrict__ whh32, // 3072 x 384 dwords (bf16 pairs)
    float* __restrict__ out,                // T x outStride, cols [colOff, +768)
    int outStride, int colOff,
    float* __restrict__ hg,                 // 2 x 768 broadcast buffers
    unsigned int* __restrict__ sync,        // pre-zeroed counter
    float* __restrict__ hn, float* __restrict__ cn,
    int reverse) {
  extern __shared__ unsigned char smem[];
  unsigned int* lds_w = (unsigned int*)smem;                     // 192 x KWPAD
  float* lds_h = (float*)(smem + (size_t)GSLICE * KWPAD * 4);    // 768 floats
  float* gbuf  = lds_h + H_DIM;                                  // 192 floats

  const int tid = threadIdx.x;
  const int hbase = blockIdx.x * HSLICE;

  // Stage this WG's 192 weight rows (resident for all 8192 steps).
  for (int idx = tid; idx < GSLICE * KW32; idx += 256) {
    const int lr = idx / KW32;
    const int kk = idx - lr * KW32;
    const int q = lr / HSLICE;
    const int rr = lr - q * HSLICE;
    const int grow = q * H_DIM + hbase + rr;
    lds_w[lr * KWPAD + kk] = whh32[(size_t)grow * KW32 + kk];
  }
  if (tid < HSLICE)
    __hip_atomic_store(&hg[hbase + tid], 0.f, __ATOMIC_RELAXED,
                       __HIP_MEMORY_SCOPE_AGENT);
  unsigned int phase = 0;
  grid_sync(sync, &phase, tid);   // weights + h0 published

  float c_reg = 0.f;
  int rb = 0;
  for (int s = 0; s < T_LEN; ++s) {
    const int t = reverse ? (T_LEN - 1 - s) : s;
    const float* __restrict__ hsrc = hg + rb * H_DIM;
    float* __restrict__ hdst = hg + (rb ^ 1) * H_DIM;

    for (int k = tid; k < H_DIM; k += 256)
      lds_h[k] = __hip_atomic_load(&hsrc[k], __ATOMIC_RELAXED,
                                   __HIP_MEMORY_SCOPE_AGENT);
    __syncthreads();

    // 192 dot-products of length 768: ds_load_b64 weights (conflict-free),
    // ds_load_b128 broadcast of h.
    if (tid < GSLICE) {
      const unsigned int* __restrict__ wr = lds_w + tid * KWPAD;
      float acc = 0.f;
      for (int j = 0; j < H_DIM / 4; ++j) {
        const uint2 w2 = *(const uint2*)(wr + 2 * j);
        const float4 hv = *(const float4*)(lds_h + 4 * j);
        acc = fmaf(bf_lo(w2.x), hv.x, acc);
        acc = fmaf(bf_hi(w2.x), hv.y, acc);
        acc = fmaf(bf_lo(w2.y), hv.z, acc);
        acc = fmaf(bf_hi(w2.y), hv.w, acc);
      }
      const int q = tid / HSLICE;
      const int rr = tid - q * HSLICE;
      const int grow = q * H_DIM + hbase + rr;
      float g = acc + pre[(size_t)t * G4 + grow];
      g = (q == 2) ? tanhf(g) : (1.f / (1.f + __expf(-g)));
      gbuf[tid] = g;
    }
    __syncthreads();

    if (tid < HSLICE) {
      const float iv = gbuf[tid];
      const float fv = gbuf[HSLICE + tid];
      const float gv = gbuf[2 * HSLICE + tid];
      const float ov = gbuf[3 * HSLICE + tid];
      c_reg = fv * c_reg + iv * gv;
      const float h = ov * tanhf(c_reg);
      out[(size_t)t * outStride + colOff + hbase + tid] = h;
      __hip_atomic_store(&hdst[hbase + tid], h, __ATOMIC_RELAXED,
                         __HIP_MEMORY_SCOPE_AGENT);
    }
    grid_sync(sync, &phase, tid);
    rb ^= 1;
  }

  if (tid < HSLICE) {
    hn[hbase + tid] = __hip_atomic_load(&hg[rb * H_DIM + hbase + tid],
                                        __ATOMIC_RELAXED, __HIP_MEMORY_SCOPE_AGENT);
    cn[hbase + tid] = c_reg;
  }
}

// --------------------------------------------------------------------- epilogue
// comb_seg is sorted in the reference, so each group's entries are a contiguous
// range: one block per group, binary-search the range, private sum (no atomics).
__global__ __launch_bounds__(256) void seg_mean_sorted(
    const float* __restrict__ out, const int* __restrict__ idx,
    const int* __restrict__ seg, float* __restrict__ dst) {
  const int g = blockIdx.x;
  __shared__ int slo, shi;
  if (threadIdx.x == 0) {
    int lo = 0, hi = E_CNT;
    while (lo < hi) { int mid = (lo + hi) >> 1; if (seg[mid] < g) lo = mid + 1; else hi = mid; }
    slo = lo;
    int lo2 = lo, hi2 = E_CNT;
    while (lo2 < hi2) { int mid = (lo2 + hi2) >> 1; if (seg[mid] < g + 1) lo2 = mid + 1; else hi2 = mid; }
    shi = lo2;
  }
  __syncthreads();
  const int lo = slo, hi = shi;
  const float inv = (hi > lo) ? 1.f / (float)(hi - lo) : 0.f;  // empty -> 0 (sum/max(cnt,1))
  for (int c = threadIdx.x; c < X1_DIM; c += 256) {
    float s = 0.f;
    for (int e = lo; e < hi; ++e) s += out[(size_t)idx[e] * X1_DIM + c];
    dst[(size_t)g * X1_DIM + c] = s * inv;
  }
}

__global__ void sent_rep_k(const float* __restrict__ out,
                           const int* __restrict__ sep, float* dst) {
  const int i = blockIdx.x;           // 0..S-1
  const int s = sep[i];
  const int e = sep[i + 1];
  for (int c = threadIdx.x; c < H_DIM; c += 256) {
    dst[(size_t)i * X1_DIM + c]         = out[(size_t)e * X1_DIM + c];
    dst[(size_t)i * X1_DIM + H_DIM + c] = out[(size_t)s * X1_DIM + H_DIM + c];
  }
}

// =============================================================== host launcher
extern "C" void kernel_launch(void* const* d_in, const int* in_sizes, int n_in,
                              void* d_out, int out_size, void* d_ws, size_t ws_size,
                              hipStream_t stream) {
  const float* x     = (const float*)d_in[0];
  const float* wih0  = (const float*)d_in[1];
  const float* whh0  = (const float*)d_in[2];
  const float* b0    = (const float*)d_in[3];
  const float* wih1  = (const float*)d_in[4];
  const float* whh1  = (const float*)d_in[5];
  const float* b1    = (const float*)d_in[6];
  const int*   cidx  = (const int*)d_in[7];
  const int*   cseg  = (const int*)d_in[8];
  const int*   sep   = (const int*)d_in[9];

  float* outf = (float*)d_out;
  const size_t OUT_OFF  = 0;
  const size_t SENT_OFF = (size_t)T_LEN * X1_DIM;
  const size_t COPY_OFF = SENT_OFF + (size_t)S_CNT * X1_DIM;
  const size_t HN_OFF   = COPY_OFF + (size_t)G_CNT * X1_DIM;
  const size_t CN_OFF   = HN_OFF + 4 * (size_t)H_DIM;

  uint8_t* ws = (uint8_t*)d_ws;
  size_t off = 0;
  auto carve = [&](size_t bytes) -> uint8_t* {
    uint8_t* p = ws + off;
    off = (off + bytes + 255) & ~(size_t)255;
    return p;
  };
  unsigned short* x_bf    = (unsigned short*)carve((size_t)T_LEN * D_IN * 2);
  unsigned short* wih0_bf = (unsigned short*)carve((size_t)2 * G4 * D_IN * 2);
  unsigned short* wih1_bf = (unsigned short*)carve((size_t)2 * G4 * X1_DIM * 2);
  unsigned short* whh_bf  = (unsigned short*)carve((size_t)4 * G4 * H_DIM * 2);
  float*          x1f     = (float*)carve((size_t)T_LEN * X1_DIM * 4);
  unsigned short* x1_bf   = (unsigned short*)carve((size_t)T_LEN * X1_DIM * 2);
  float*          pre     = (float*)carve((size_t)T_LEN * G4 * 4);
  float*          hg      = (float*)carve(2 * (size_t)H_DIM * 4);
  unsigned int*   syncc   = (unsigned int*)carve(256);

  const int THR = 256;
  auto grid1 = [&](int n) { return dim3((n + THR - 1) / THR); };

  { int n = T_LEN * D_IN;    cvt_f32_bf16<<<grid1(n), THR, 0, stream>>>(x, x_bf, n); }
  { int n = 2 * G4 * D_IN;   cvt_f32_bf16<<<grid1(n), THR, 0, stream>>>(wih0, wih0_bf, n); }
  { int n = 2 * G4 * X1_DIM; cvt_f32_bf16<<<grid1(n), THR, 0, stream>>>(wih1, wih1_bf, n); }
  { int n = 2 * G4 * H_DIM;  cvt_f32_bf16<<<grid1(n), THR, 0, stream>>>(whh0, whh_bf, n); }
  { int n = 2 * G4 * H_DIM;  cvt_f32_bf16<<<grid1(n), THR, 0, stream>>>(whh1, whh_bf + (size_t)2 * G4 * H_DIM, n); }

  const int gemmBlocks = (T_LEN / 128) * (G4 / 64);                       // 3072
  const size_t SMEM = (size_t)GSLICE * KWPAD * 4 + H_DIM * 4 + GSLICE * 4; // 300,288 B

  auto run_dir = [&](const unsigned short* A, int K,
                     const unsigned short* wihD, const float* biasD,
                     const unsigned int* whhD,
                     float* outBase, int colOff, int hnRow, int reverse) {
    gemm_bf16_wmma<<<gemmBlocks, THR, 0, stream>>>(A, wihD, biasD, pre,
                                                   T_LEN, G4, K);
    zero_u32<<<1, 64, 0, stream>>>(syncc, 64);
    lstm_recur<<<NWG_REC, THR, SMEM, stream>>>(
        pre, whhD, outBase, X1_DIM, colOff, hg, syncc,
        outf + HN_OFF + (size_t)hnRow * H_DIM,
        outf + CN_OFF + (size_t)hnRow * H_DIM, reverse);
  };

  const unsigned int* whh32 = (const unsigned int*)whh_bf;
  const size_t whhSlice = (size_t)G4 * H_DIM / 2;   // dwords per (layer,dir) slice

  // layer 0 (K = 512)
  run_dir(x_bf, D_IN, wih0_bf,                     b0,      whh32 + 0 * whhSlice, x1f, 0,     0, 0);
  run_dir(x_bf, D_IN, wih0_bf + (size_t)G4 * D_IN, b0 + G4, whh32 + 1 * whhSlice, x1f, H_DIM, 1, 1);

  // layer 1 (K = 1536)
  { int n = T_LEN * X1_DIM; cvt_f32_bf16<<<grid1(n), THR, 0, stream>>>(x1f, x1_bf, n); }
  run_dir(x1_bf, X1_DIM, wih1_bf,                       b1,      whh32 + 2 * whhSlice, outf + OUT_OFF, 0,     2, 0);
  run_dir(x1_bf, X1_DIM, wih1_bf + (size_t)G4 * X1_DIM, b1 + G4, whh32 + 3 * whhSlice, outf + OUT_OFF, H_DIM, 3, 1);

  // copy_rep (sorted segments -> atomic-free mean) and sent_rep
  seg_mean_sorted<<<G_CNT, THR, 0, stream>>>(outf + OUT_OFF, cidx, cseg, outf + COPY_OFF);
  sent_rep_k<<<S_CNT, THR, 0, stream>>>(outf + OUT_OFF, sep, outf + SENT_OFF);
}
